// PhysicsNODE_78331613545127
// MI455X (gfx1250) — matrix operand order repaired and checked by reading, not compiled, not run on hardware
//
#include <hip/hip_runtime.h>
#include <hip/hip_bf16.h>
#include <math.h>

typedef __attribute__((ext_vector_type(16))) _Float16 v16h;
typedef __attribute__((ext_vector_type(8)))  _Float16 v8h;
typedef __attribute__((ext_vector_type(8)))  float    v8f;
typedef __attribute__((ext_vector_type(4)))  float    v4f;

union Frag { v16h v; v8h h2[2]; _Float16 e[16]; };

#define EPS_NUC  1e-4f
#define F_INIT   1e-3f
#define F_EQ_TOL 0.999f

// fast hardware reciprocal (v_rcp_f32) based activations -- avoids the IEEE
// div expansion (v_div_scale/v_div_fmas chains) seen in the previous disasm.
__device__ __forceinline__ float rcp_f(float x) { return __builtin_amdgcn_rcpf(x); }
__device__ __forceinline__ float silu_f(float x) {
  return x * rcp_f(1.0f + __expf(-x));
}
__device__ __forceinline__ float tanh_f(float x) {
  float t = __expf(-2.0f * fabsf(x));          // in (0,1]; never overflows
  float r = (1.0f - t) * rcp_f(1.0f + t);
  return copysignf(r, x);
}
__device__ __forceinline__ float softplus_f(float x) { return (x > 20.0f) ? x : log1pf(__expf(x)); }

// ---------------------------------------------------------------------------
// Kernel 1: conditioning network. One wave (32 lanes) per batch element.
// lane = embedding channel. Produces cond[37] (padded to 40 f32) in d_ws.
// ---------------------------------------------------------------------------
__global__ __launch_bounds__(256) void cond_kernel(
    const float* __restrict__ sf, const float* __restrict__ feq, const float* __restrict__ dG,
    const float* __restrict__ cew,  const float* __restrict__ ceb,  const float* __restrict__ cel,
    const float* __restrict__ wqkv, const float* __restrict__ bqkv,
    const float* __restrict__ wout, const float* __restrict__ bout,
    const float* __restrict__ lng,  const float* __restrict__ lnb,
    const float* __restrict__ cpw,  const float* __restrict__ cpb,
    const float* __restrict__ pw1,  const float* __restrict__ pb1,
    const float* __restrict__ pw2,  const float* __restrict__ pb2,
    float* __restrict__ condp)
{
  __shared__ float s_tok[8][5][32];
  __shared__ float s_qkv[8][5][96];
  __shared__ float s_o  [8][5][32];
  __shared__ float s_pool[8][32];
  __shared__ float s_ph [8][32];

  const int w = threadIdx.x >> 5;
  const int e = threadIdx.x & 31;
  const int b = blockIdx.x * 8 + w;

  float comp[5];
  #pragma unroll
  for (int t = 0; t < 5; t++) comp[t] = sf[b * 8 + t];
  #pragma unroll
  for (int t = 0; t < 5; t++)
    s_tok[w][t][e] = comp[t] * cew[e] + ceb[e] + cel[t * 32 + e];
  __syncthreads();

  // qkv = tokens @ W(32x96) + b : each lane computes 3 columns per token
  for (int t = 0; t < 5; t++)
    for (int part = 0; part < 3; part++) {
      int c = part * 32 + e;
      float acc = bqkv[c];
      for (int j = 0; j < 32; j++) acc += s_tok[w][t][j] * wqkv[j * 96 + c];
      s_qkv[w][t][c] = acc;
    }
  __syncthreads();

  // attention (4 heads, dim 8); lane's output channel e belongs to head e>>3
  {
    const int qb = (e >> 3) * 8;
    const float scale = 0.35355339059327373f; // 1/sqrt(8)
    for (int t = 0; t < 5; t++) {
      float sc[5], m = -1e30f;
      for (int k = 0; k < 5; k++) {
        float s = 0.0f;
        for (int d = 0; d < 8; d++) s += s_qkv[w][t][qb + d] * s_qkv[w][k][32 + qb + d];
        sc[k] = s * scale; m = fmaxf(m, sc[k]);
      }
      float den = 0.0f;
      for (int k = 0; k < 5; k++) { sc[k] = __expf(sc[k] - m); den += sc[k]; }
      float o = 0.0f;
      for (int k = 0; k < 5; k++) o += sc[k] * s_qkv[w][k][64 + e];
      s_o[w][t][e] = o * rcp_f(den);
    }
  }
  __syncthreads();

  // out projection + mean pool over tokens
  float pooled = 0.0f;
  for (int t = 0; t < 5; t++) {
    float acc = bout[e];
    for (int j = 0; j < 32; j++) acc += s_o[w][t][j] * wout[j * 32 + e];
    pooled += acc;
  }
  pooled *= 0.2f;

  // layernorm across 32 channels via wave32 shuffles
  float mu = pooled;
  for (int m = 16; m >= 1; m >>= 1) mu += __shfl_xor(mu, m, 32);
  mu *= (1.0f / 32.0f);
  float d = pooled - mu;
  float var = d * d;
  for (int m = 16; m >= 1; m >>= 1) var += __shfl_xor(var, m, 32);
  var *= (1.0f / 32.0f);
  s_pool[w][e] = d * rsqrtf(var + 1e-5f) * lng[e] + lnb[e];

  // phys MLP hidden
  float pin[5];
  pin[0] = sf[b * 8 + 5]; pin[1] = sf[b * 8 + 6]; pin[2] = sf[b * 8 + 7];
  pin[3] = feq[b];        pin[4] = dG[b];
  float ph = pb1[e];
  #pragma unroll
  for (int i = 0; i < 5; i++) ph += pin[i] * pw1[i * 32 + e];
  s_ph[w][e] = silu_f(ph);
  __syncthreads();

  // comp_embed = pooled_ln @ ce_proj
  float ce2 = cpb[e];
  for (int j = 0; j < 32; j++) ce2 += s_pool[w][j] * cpw[j * 32 + e];
  condp[b * 40 + e] = ce2;

  if (e < 8) {
    float po = 0.0f;
    if (e < 5) {
      po = pb2[e];
      for (int j = 0; j < 32; j++) po += s_ph[w][j] * pw2[j * 5 + e];
    }
    condp[b * 40 + 32 + e] = po; // cols 37..39 are zero padding
  }
}

// ---------------------------------------------------------------------------
// Kernel 2: RK4 neural ODE. One workgroup (16 wave32) per 16 batch rows.
// Wave w owns hidden columns [16w, 16w+16): W1/W2 B-fragments live in VGPRs,
// FiLM gains/biases live in VGPRs, activations bounce through LDS as f16.
// ---------------------------------------------------------------------------
__global__ __launch_bounds__(512) void ode_kernel(
    const float* __restrict__ feq,  const float* __restrict__ dG,  const float* __restrict__ tsg,
    const float* __restrict__ w1g,  const float* __restrict__ b1g,
    const float* __restrict__ w2g,  const float* __restrict__ b2g,
    const float* __restrict__ f1gw, const float* __restrict__ f1gb,
    const float* __restrict__ f1bw, const float* __restrict__ f1bb,
    const float* __restrict__ f2gw, const float* __restrict__ f2gb,
    const float* __restrict__ f2bw, const float* __restrict__ f2bb,
    const float* __restrict__ ratew, const float* __restrict__ rateb,
    const float* __restrict__ augw,  const float* __restrict__ augb,
    const float* __restrict__ condp, float* __restrict__ outp, int T)
{
  // strides 72 / 264 halfs: rows stay 16B-aligned and LDS banks de-conflicted
  __shared__ __align__(16) _Float16 xb [16 * 72];
  __shared__ __align__(16) _Float16 h1b[16 * 264];
  __shared__ __align__(16) _Float16 h2b[16 * 264];
  __shared__ float condL[16 * 40];
  __shared__ __align__(16) float w3t[9 * 260];   // head weights, [j][256] rows
  __shared__ float hbias[9];
  __shared__ float yst[144], ytmp[144], kst[4][144];
  __shared__ float tsL[64];
  __shared__ float driveL[16], feqcL[16], feqtolL[16];

  const int tid  = threadIdx.x;
  const int b0   = blockIdx.x * 16;
  const int wv   = tid >> 5;
  const int lane = tid & 31;
  const int lo   = lane & 15;
  const int hi   = lane >> 4;
  const int colBase = wv * 16;
  const int col  = colBase + lo;

  // pull the big resident weights toward this WGP (global_prefetch_b8)
  __builtin_prefetch(w2g + col, 0, 3);
  __builtin_prefetch(w2g + 128 * 256 + col, 0, 3);

  // ---- one-time LDS init ----
  for (int i = tid; i < 16 * 72; i += 512) xb[i] = (_Float16)0.0f;
  for (int i = tid; i < 16 * 40; i += 512) condL[i] = condp[b0 * 40 + i];
  for (int i = tid; i < 9 * 256; i += 512) {
    int j = i / 256, k = i % 256;
    w3t[j * 260 + k] = (j == 0) ? ratew[k] : augw[k * 8 + (j - 1)];
  }
  if (tid < 9)  hbias[tid] = (tid == 0) ? rateb[0] : augb[tid - 1];
  if (tid < T)  tsL[tid] = tsg[tid];
  if (tid < 16) {
    float fe = feq[b0 + tid];
    driveL[tid]  = fabsf(dG[b0 + tid]) + 0.1f;
    feqcL[tid]   = fmaxf(fe, 1e-6f);
    feqtolL[tid] = fe * F_EQ_TOL;
  }
  if (tid < 144) yst[tid] = (tid % 9 == 0) ? F_INIT : 0.0f;
  __syncthreads();

  // constant part of x tile: cols 9..45 = cond, 46..71 stay zero
  for (int i = tid; i < 16 * 37; i += 512) {
    int r = i / 37, c = i % 37;
    xb[r * 72 + 9 + c] = (_Float16)condL[r * 40 + c];
  }
  __syncthreads();

  // ---- FiLM gains/biases into registers (lane's C-tile positions) ----
  float g1r[8], b1r[8], g2r[8], b2r[8];
  #pragma unroll
  for (int v = 0; v < 8; v++) {
    int r = v + hi * 8;
    float a1 = f1gb[col], a2 = f1bb[col], a3 = f2gb[col], a4 = f2bb[col];
    for (int i2 = 0; i2 < 37; i2++) {
      float cv = condL[r * 40 + i2];
      a1 += cv * f1gw[i2 * 256 + col];
      a2 += cv * f1bw[i2 * 256 + col];
      a3 += cv * f2gw[i2 * 256 + col];
      a4 += cv * f2bw[i2 * 256 + col];
    }
    g1r[v] = 1.0f + a1; b1r[v] = a2; g2r[v] = 1.0f + a3; b2r[v] = a4;
  }
  const float bn1 = b1g[col], bn2 = b2g[col];

  // ---- resident weight B-fragments (K x 16 per wave, f16) ----
  // B layout: lanes 0-15 hold K=k0..k0+7,k0+16..k0+23 of column `lo`; lanes 16-31 hold K+8.
  Frag w1f[2], w2f[8];
  #pragma unroll
  for (int ck = 0; ck < 2; ck++) {
    int k0 = ck * 32 + hi * 8;
    #pragma unroll
    for (int i = 0; i < 16; i++) {
      int k = k0 + i + ((i >= 8) ? 8 : 0);
      w1f[ck].e[i] = (_Float16)((k < 46) ? w1g[k * 256 + col] : 0.0f);
    }
  }
  #pragma unroll
  for (int ck = 0; ck < 8; ck++) {
    int k0 = ck * 32 + hi * 8;
    #pragma unroll
    for (int i = 0; i < 16; i++)
      w2f[ck].e[i] = (_Float16)w2g[(k0 + i + ((i >= 8) ? 8 : 0)) * 256 + col];
  }

  // A-fragment loader per documented 16-bit 16x32 layout:
  // lanes 0-15 (row=lo): K = c0..c0+7 then c0+16..c0+23
  // lanes 16-31(row=lo): K = c0+8..c0+15 then c0+24..c0+31
  auto load_afrag = [&](const _Float16* base, int strideH, int c0) -> v16h {
    const _Float16* p = base + lo * strideH + c0 + hi * 8;
    Frag f;
    f.h2[0] = *(const v8h*)p;
    f.h2[1] = *(const v8h*)(p + 16);
    return f.v;
  };

  const v16h a1frag = load_afrag(xb, 72, 32); // pure-cond half, constant all run

  if (tid < 16)
    outp[(b0 + tid) * T + 0] = fminf(fmaxf(F_INIT, 0.0f), feqtolL[tid]);

  const float coef[4] = {0.0f, 0.5f, 0.5f, 1.0f};
  for (int step = 0; step < T - 1; step++) {
    const float dt = tsL[step + 1] - tsL[step];
    for (int s = 0; s < 4; s++) {
      // stage input y_temp -> x tile (only first 9 columns change)
      if (tid < 144) {
        int r = tid / 9, c = tid % 9;
        float yt = yst[tid];
        if (s > 0) yt += coef[s] * dt * kst[s - 1][tid];
        ytmp[tid] = yt;
        xb[r * 72 + c] = (_Float16)yt;
      }
      __syncthreads();

      // layer 1: [16x64] @ [64x16] per wave
      v16h a0 = load_afrag(xb, 72, 0);
      v8f acc = {0.f, 0.f, 0.f, 0.f, 0.f, 0.f, 0.f, 0.f};
      acc = __builtin_amdgcn_wmma_f32_16x16x32_f16(false, a0,     false, w1f[0].v, (short)0, acc, false, false);
      acc = __builtin_amdgcn_wmma_f32_16x16x32_f16(false, a1frag, false, w1f[1].v, (short)0, acc, false, false);
      #pragma unroll
      for (int v = 0; v < 8; v++) {
        int r = v + hi * 8;
        float h = silu_f(acc[v] + bn1) * g1r[v] + b1r[v];
        h1b[r * 264 + col] = (_Float16)h;
      }
      __syncthreads();

      // layer 2: [16x256] @ [256x16] per wave
      v8f acc2 = {0.f, 0.f, 0.f, 0.f, 0.f, 0.f, 0.f, 0.f};
      #pragma unroll
      for (int ck = 0; ck < 8; ck++) {
        v16h a = load_afrag(h1b, 264, ck * 32);
        acc2 = __builtin_amdgcn_wmma_f32_16x16x32_f16(false, a, false, w2f[ck].v, (short)0, acc2, false, false);
      }
      #pragma unroll
      for (int v = 0; v < 8; v++) {
        int r = v + hi * 8;
        float h = silu_f(acc2[v] + bn2) * g2r[v] + b2r[v];
        h2b[r * 264 + col] = (_Float16)h;
      }
      __syncthreads();

      // heads: 16 rows x 9 outputs, K=256, vectorized b128 LDS traffic
      if (tid < 144) {
        int r = tid / 9, j = tid % 9;
        const _Float16* hp = &h2b[r * 264];
        const float*     wp = &w3t[j * 260];
        float z = hbias[j];
        #pragma unroll 4
        for (int k = 0; k < 256; k += 8) {
          v8h hv = *(const v8h*)(hp + k);
          v4f w0 = *(const v4f*)(wp + k);
          v4f w1 = *(const v4f*)(wp + k + 4);
          z += (float)hv[0] * w0[0] + (float)hv[1] * w0[1] +
               (float)hv[2] * w0[2] + (float)hv[3] * w0[3] +
               (float)hv[4] * w1[0] + (float)hv[5] * w1[1] +
               (float)hv[6] * w1[2] + (float)hv[7] * w1[3];
        }
        float kv;
        if (j == 0) {
          float rate = softplus_f(z) * 0.01f;
          float fnow = fmaxf(ytmp[r * 9], 0.0f);
          kv = rate * driveL[r] * (fmaxf(feqcL[r] - fnow, 0.0f) + 1e-6f) * (fnow + EPS_NUC);
        } else {
          kv = 0.1f * tanh_f(z);
        }
        kst[s][tid] = kv;
      }
      __syncthreads();
    }

    if (tid < 144)
      yst[tid] += (dt * (1.0f / 6.0f)) *
                  (kst[0][tid] + 2.0f * kst[1][tid] + 2.0f * kst[2][tid] + kst[3][tid]);
    __syncthreads();
    if (tid < 16)
      outp[(b0 + tid) * T + step + 1] = fminf(fmaxf(yst[tid * 9], 0.0f), feqtolL[tid]);
    __syncthreads();
  }
}

// ---------------------------------------------------------------------------
extern "C" void kernel_launch(void* const* d_in, const int* in_sizes, int n_in,
                              void* d_out, int out_size, void* d_ws, size_t ws_size,
                              hipStream_t stream)
{
  const float* sf  = (const float*)d_in[0];
  const float* feq = (const float*)d_in[1];
  const float* dG  = (const float*)d_in[2];
  const float* ts  = (const float*)d_in[3];
  const float* P[31];
  for (int i = 0; i < 31; i++) P[i] = (const float*)d_in[4 + i];
  // P: 0 ce_embed_w 1 ce_embed_b 2 ce_elem 3 attn_in_w 4 attn_in_b 5 attn_out_w
  //    6 attn_out_b 7 ln_g 8 ln_b 9 ce_proj_w 10 ce_proj_b 11 pp_w1 12 pp_b1
  //    13 pp_w2 14 pp_b2 15 net_w1 16 net_b1 17 net_w2 18 net_b2 19 f1g_w
  //    20 f1g_b 21 f1b_w 22 f1b_b 23 f2g_w 24 f2g_b 25 f2b_w 26 f2b_b
  //    27 rate_w 28 rate_b 29 aug_w 30 aug_b

  const int B = in_sizes[1];   // 16384
  const int T = in_sizes[3];   // 33
  float* condp = (float*)d_ws; // B x 40 f32

  cond_kernel<<<B / 8, 256, 0, stream>>>(
      sf, feq, dG,
      P[0], P[1], P[2], P[3], P[4], P[5], P[6], P[7], P[8],
      P[9], P[10], P[11], P[12], P[13], P[14],
      condp);

  ode_kernel<<<B / 16, 512, 0, stream>>>(
      feq, dG, ts,
      P[15], P[16], P[17], P[18],
      P[19], P[20], P[21], P[22], P[23], P[24], P[25], P[26],
      P[27], P[28], P[29], P[30],
      condp, (float*)d_out, T);
}
